// Attention_58471684767960
// MI455X (gfx1250) — compile-verified, matrix operand order checked by reading, not run on hardware
//
#include <hip/hip_runtime.h>
#include <hip/hip_bf16.h>
#include <math.h>

// Problem constants (fixed by the reference)
#define BA   4
#define SEQ  2048
#define DM   1024
#define NH   16
#define HD   64
#define MROWS (BA * SEQ)   // 8192

typedef __bf16 bf16;
typedef __attribute__((ext_vector_type(16))) __bf16 v16bf;
typedef __attribute__((ext_vector_type(8)))  __bf16 v8bf;
typedef __attribute__((ext_vector_type(8)))  float  v8f;

union frag16 { v16bf v; struct { v8bf lo, hi; } h; };

// Fragment load for the CDNA5 16-bit 16x32 A layout (and B stored
// column-contiguous so the same math applies):
//   lane L: row = L%16 ; halves 0..7 -> k = (L>=16?8:0)+0..7
//                        halves 8..15 -> k = 16+(L>=16?8:0)+0..7
// Two contiguous 16-byte loads per lane.
__device__ __forceinline__ v16bf load_frag(const bf16* tile, int ld) {
    const int lane = threadIdx.x & 31;
    const bf16* p = tile + (size_t)(lane & 15) * ld + ((lane >> 4) << 3);
    frag16 f;
    f.h.lo = *(const v8bf*)(p);
    f.h.hi = *(const v8bf*)(p + 16);
    return f.v;
}

__device__ __forceinline__ v8f wmma_bf16(v16bf a, v16bf b, v8f c) {
    // D = A(16x32 bf16) * B(32x16 bf16) + C(16x16 f32)
    return __builtin_amdgcn_wmma_f32_16x16x32_bf16(
        /*neg_a=*/false, a, /*neg_b=*/false, b,
        /*c_mod=*/(short)0, c, /*reuse_a=*/false, /*reuse_b=*/false);
}

// ---------------------------------------------------------------- convert X
__global__ __launch_bounds__(256) void cvt_f32_bf16_kernel(
        const float* __restrict__ x, bf16* __restrict__ y, int n) {
    int idx = (blockIdx.x * blockDim.x + threadIdx.x) * 4;
    int stride = gridDim.x * blockDim.x * 4;
    for (int i = idx; i < n; i += stride) {
        float4 v = *(const float4*)(x + i);
        y[i + 0] = (bf16)v.x;
        y[i + 1] = (bf16)v.y;
        y[i + 2] = (bf16)v.z;
        y[i + 3] = (bf16)v.w;
    }
}

// --------------------------------------------- transpose W[k][n] -> Wt[n][k]
__global__ __launch_bounds__(256) void transpose_w_kernel(
        const float* __restrict__ W, bf16* __restrict__ Wt) {
    __shared__ float tile[32][33];
    const int bx = blockIdx.x * 32;   // n tile
    const int by = blockIdx.y * 32;   // k tile
    const int tx = threadIdx.x & 31;
    const int ty = threadIdx.x >> 5;  // 0..7
    #pragma unroll
    for (int i = ty; i < 32; i += 8)
        tile[i][tx] = W[(size_t)(by + i) * DM + bx + tx];
    __syncthreads();
    #pragma unroll
    for (int i = ty; i < 32; i += 8)
        Wt[(size_t)(bx + i) * DM + by + tx] = (bf16)tile[tx][i];
}

// ------------------------------------------------------------ WMMA GEMM
// C[M=8192, N=1024] = A[M,1024] @ W[1024,N] + bias, W given transposed (Wt[N][K]).
// Workgroup: 4 waves -> 64 rows x 64 cols; each wave does a 16x64 strip.
// mode 0: bf16 out, [B,H,S,DK] head-split layout   (Q, K)
// mode 1: bf16 out, [B,H,DK,S] transposed layout   (V)
// mode 2: f32  out, row-major [M,N]                (final projection)
__global__ __launch_bounds__(128) void gemm_bf16_kernel(
        const bf16* __restrict__ A, const bf16* __restrict__ Wt,
        const float* __restrict__ bias, bf16* __restrict__ outb,
        float* __restrict__ outf, int mode) {
    const int wave = threadIdx.x >> 5;
    const int lane = threadIdx.x & 31;
    const int m0 = blockIdx.x * 64 + wave * 16;
    const int n0 = blockIdx.y * 64;

    v8f acc[4] = {{}, {}, {}, {}};
    for (int k = 0; k < DM; k += 32) {
        v16bf a = load_frag(A + (size_t)m0 * DM + k, DM);
        #pragma unroll
        for (int nt = 0; nt < 4; ++nt) {
            v16bf b = load_frag(Wt + (size_t)(n0 + nt * 16) * DM + k, DM);
            acc[nt] = wmma_bf16(a, b, acc[nt]);
        }
    }

    const int rbase = (lane >> 4) << 3;   // row offset within tile: 0 or 8
    #pragma unroll
    for (int nt = 0; nt < 4; ++nt) {
        const int n = n0 + nt * 16 + (lane & 15);
        const float bv = bias[n];
        #pragma unroll
        for (int r = 0; r < 8; ++r) {
            const int m = m0 + rbase + r;
            const float v = acc[nt][r] + bv;
            if (mode == 2) {
                outf[(size_t)m * DM + n] = v;
            } else {
                const int b = m >> 11;          // m / SEQ
                const int s = m & (SEQ - 1);
                const int h = n >> 6;           // n / HD
                const int dk = n & (HD - 1);
                if (mode == 0)   // [B,H,S,DK]
                    outb[(((size_t)b * NH + h) * SEQ + s) * HD + dk] = (bf16)v;
                else             // [B,H,DK,S]
                    outb[(((size_t)b * NH + h) * HD + dk) * SEQ + s] = (bf16)v;
            }
        }
    }
}

// ------------------------------------------------------------ attention core
// One workgroup (4 waves, 128 threads) per (batch*head, 16-row q tile).
// Full 16xSEQ f32 score strip + 16xSEQ bf16 P strip live in the 320KB WGP LDS,
// so the softmax is exact with a single pass over K and a single pass over V.
#define ATT_LDS_BYTES (16 * SEQ * 4 + 16 * SEQ * 2 + (128 + 16 + 16) * 4)

__global__ __launch_bounds__(128) void attention_kernel(
        const bf16* __restrict__ Qh,   // [B,H,S,DK] bf16
        const bf16* __restrict__ Kh,   // [B,H,S,DK] bf16
        const bf16* __restrict__ Vt,   // [B,H,DK,S] bf16
        bf16* __restrict__ Zb) {       // [M, D]     bf16 (heads recombined)
    extern __shared__ char smem[];
    float* sS     = (float*)smem;                          // [16][SEQ] scores
    bf16*  sP     = (bf16*)(smem + 16 * SEQ * 4);          // [16][SEQ] exp(s-max)
    float* red    = (float*)(smem + 16 * SEQ * 4 + 16 * SEQ * 2); // [16][8]
    float* rowmax = red + 128;                             // [16]
    float* rowsum = rowmax + 16;                           // [16]

    const int wave = threadIdx.x >> 5;
    const int lane = threadIdx.x & 31;
    const int q0 = blockIdx.x * 16;
    const int bh = blockIdx.y;               // b*NH + h
    const int rbase = (lane >> 4) << 3;

    // ---- phase 1: scores = Q @ K^T / sqrt(DK), waves split the kv range
    const bf16* Qt = Qh + ((size_t)bh * SEQ + q0) * HD;
    const v16bf qa0 = load_frag(Qt + 0, HD);
    const v16bf qa1 = load_frag(Qt + 32, HD);

    const int kvBeg = wave * (SEQ / 4);
    const int kvEnd = kvBeg + (SEQ / 4);
    for (int kt = kvBeg; kt < kvEnd; kt += 16) {
        const bf16* Kt = Kh + ((size_t)bh * SEQ + kt) * HD;
        v8f acc = {};
        acc = wmma_bf16(qa0, load_frag(Kt + 0, HD), acc);
        acc = wmma_bf16(qa1, load_frag(Kt + 32, HD), acc);
        const int col = kt + (lane & 15);
        #pragma unroll
        for (int r = 0; r < 8; ++r)
            sS[(size_t)(rbase + r) * SEQ + col] = acc[r] * 0.125f; // 1/sqrt(64)
    }
    __syncthreads();

    // ---- phase 2: exact softmax over each of the 16 rows (8 threads/row)
    const int row = threadIdx.x >> 3;
    const int seg = threadIdx.x & 7;
    float* srow = sS + (size_t)row * SEQ;
    const int c0 = seg * (SEQ / 8), c1 = c0 + (SEQ / 8);

    float mx = -1e30f;
    for (int c = c0; c < c1; ++c) mx = fmaxf(mx, srow[c]);
    red[row * 8 + seg] = mx;
    __syncthreads();
    if (seg == 0) {
        float m = red[row * 8];
        #pragma unroll
        for (int i = 1; i < 8; ++i) m = fmaxf(m, red[row * 8 + i]);
        rowmax[row] = m;
    }
    __syncthreads();

    const float rmax = rowmax[row];
    bf16* prow = sP + (size_t)row * SEQ;
    float sum = 0.f;
    for (int c = c0; c < c1; ++c) {
        const float e = __expf(srow[c] - rmax);
        sum += e;
        prow[c] = (bf16)e;                 // unnormalized P; 1/sum folded later
    }
    red[row * 8 + seg] = sum;
    __syncthreads();
    if (seg == 0) {
        float s = 0.f;
        #pragma unroll
        for (int i = 0; i < 8; ++i) s += red[row * 8 + i];
        rowsum[row] = s;
    }
    __syncthreads();

    // ---- phase 3: Z = (P / rowsum) @ V ; waves split the DK dimension
    const bf16* Vb = Vt + ((size_t)bh * HD + wave * 16) * SEQ;
    v8f acc = {};
    for (int kt = 0; kt < SEQ; kt += 32) {
        v16bf pa = load_frag(sP + kt, SEQ);    // A: 16 q rows x 32 kv (LDS)
        v16bf vb = load_frag(Vb + kt, SEQ);    // B: 32 kv x 16 dk (V^T, contig)
        acc = wmma_bf16(pa, vb, acc);
    }

    const int b = bh / NH, h = bh % NH;
    const int col = wave * 16 + (lane & 15);   // dk within head
    #pragma unroll
    for (int r = 0; r < 8; ++r) {
        const int m = rbase + r;
        const float z = acc[r] * (1.0f / rowsum[m]);
        Zb[(size_t)(b * SEQ + q0 + m) * DM + h * HD + col] = (bf16)z;
    }
}

// ---------------------------------------------------------------- launcher
extern "C" void kernel_launch(void* const* d_in, const int* in_sizes, int n_in,
                              void* d_out, int out_size, void* d_ws, size_t ws_size,
                              hipStream_t stream) {
    const float* X  = (const float*)d_in[0];
    const float* Wq = (const float*)d_in[1];
    const float* bq = (const float*)d_in[2];
    const float* Wk = (const float*)d_in[3];
    const float* bk = (const float*)d_in[4];
    const float* Wv = (const float*)d_in[5];
    const float* bv = (const float*)d_in[6];
    const float* Wo = (const float*)d_in[7];
    const float* bo = (const float*)d_in[8];

    char* ws = (char*)d_ws;
    size_t off = 0;
    auto carve = [&](size_t bytes) {
        void* p = ws + off;
        off += (bytes + 255) & ~(size_t)255;
        return p;
    };
    bf16* Xb  = (bf16*)carve((size_t)MROWS * DM * 2);  // X in bf16
    bf16* Wqt = (bf16*)carve((size_t)DM * DM * 2);     // transposed bf16 weights
    bf16* Wkt = (bf16*)carve((size_t)DM * DM * 2);
    bf16* Wvt = (bf16*)carve((size_t)DM * DM * 2);
    bf16* Wot = (bf16*)carve((size_t)DM * DM * 2);
    bf16* Qh  = (bf16*)carve((size_t)MROWS * DM * 2);  // [B,H,S,DK]
    bf16* Kh  = (bf16*)carve((size_t)MROWS * DM * 2);  // [B,H,S,DK]
    bf16* Vt  = (bf16*)carve((size_t)MROWS * DM * 2);  // [B,H,DK,S]
    bf16* Zb  = (bf16*)carve((size_t)MROWS * DM * 2);  // [M, D]

    // 1) precision conversion + weight transposes
    cvt_f32_bf16_kernel<<<8192, 256, 0, stream>>>(X, Xb, MROWS * DM);
    dim3 tgrid(DM / 32, DM / 32);
    transpose_w_kernel<<<tgrid, 256, 0, stream>>>(Wq, Wqt);
    transpose_w_kernel<<<tgrid, 256, 0, stream>>>(Wk, Wkt);
    transpose_w_kernel<<<tgrid, 256, 0, stream>>>(Wv, Wvt);
    transpose_w_kernel<<<tgrid, 256, 0, stream>>>(Wo, Wot);

    // 2) Q/K/V projections (WMMA)
    dim3 ggrid(MROWS / 64, DM / 64);
    gemm_bf16_kernel<<<ggrid, 128, 0, stream>>>(Xb, Wqt, bq, Qh, nullptr, 0);
    gemm_bf16_kernel<<<ggrid, 128, 0, stream>>>(Xb, Wkt, bk, Kh, nullptr, 0);
    gemm_bf16_kernel<<<ggrid, 128, 0, stream>>>(Xb, Wvt, bv, Vt, nullptr, 1);

    // 3) attention core (WMMA + exact softmax fully resident in 320KB LDS)
    dim3 agrid(SEQ / 16, BA * NH);
    attention_kernel<<<agrid, 128, ATT_LDS_BYTES, stream>>>(Qh, Kh, Vt, Zb);

    // 4) output projection -> f32 result
    gemm_bf16_kernel<<<ggrid, 128, 0, stream>>>(Zb, Wot, bo, nullptr,
                                                (float*)d_out, 2);
}
